// ContrastiveLoss_3478923510126
// MI455X (gfx1250) — matrix operand order repaired and checked by reading, not compiled, not run on hardware
//
#include <hip/hip_runtime.h>
#include <hip/hip_bf16.h>

typedef _Float16 h4  __attribute__((ext_vector_type(4)));
typedef _Float16 h8  __attribute__((ext_vector_type(8)));
typedef _Float16 h16 __attribute__((ext_vector_type(16)));
typedef float    f8  __attribute__((ext_vector_type(8)));

#define NPAIR   4096      // N
#define DDIM    128       // D
#define TWO_N   8192      // 2N
#define CSPLIT  8         // column splits for the fused GEMM+LSE
#define COLS_PER_SPLIT (TWO_N / CSPLIT)        // 1024
#define ITERS_PER_SPLIT (COLS_PER_SPLIT / 32)  // 32 (32 columns per iteration)
#define STAGE_BYTES (32 * DDIM * 2)            // 8192 B per 32-row B stage
// exp(2a - 2) == exp2(a*EXP2C - EXP2C)
#define EXP2C 2.885390081777926816f            // 2 * log2(e)

// ---------------------------------------------------------------------------
// Kernel 1: L2-normalize rows of [emb_i; emb_j] -> f16 reps in workspace.
// One wave32 per row; each lane handles 4 of the 128 elements.
// ---------------------------------------------------------------------------
__global__ void __launch_bounds__(256)
k_normalize(const float* __restrict__ ei, const float* __restrict__ ej,
            _Float16* __restrict__ reps) {
    const int lane = threadIdx.x & 31;
    const int wave = threadIdx.x >> 5;
    const int row  = blockIdx.x * (blockDim.x >> 5) + wave;
    if (row >= TWO_N) return;

    const float* src = (row < NPAIR) ? (ei + (size_t)row * DDIM)
                                     : (ej + (size_t)(row - NPAIR) * DDIM);
    float4 v = ((const float4*)src)[lane];
    float ss = v.x * v.x + v.y * v.y + v.z * v.z + v.w * v.w;
    ss += __shfl_xor(ss, 1);
    ss += __shfl_xor(ss, 2);
    ss += __shfl_xor(ss, 4);
    ss += __shfl_xor(ss, 8);
    ss += __shfl_xor(ss, 16);
    const float inv = rsqrtf(ss);

    h4 o;
    o[0] = (_Float16)(v.x * inv);
    o[1] = (_Float16)(v.y * inv);
    o[2] = (_Float16)(v.z * inv);
    o[3] = (_Float16)(v.w * inv);
    ((h4*)(reps + (size_t)row * DDIM))[lane] = o;
}

// ---------------------------------------------------------------------------
// Kernel 2: fused Gram-matrix (WMMA f16 -> f32) + row-wise sum of exp.
// Each wave owns a 16-row tile; grid.y splits the 8192 columns 8 ways.
// 32 columns (two 16x16 WMMA tiles) are staged into LDS per iteration with
// double-buffered async copies shared by the whole block (one barrier per
// 32 columns). No masking: the diagonal (self) term is included here and
// subtracted exactly in the finalize kernel.
// Writes sPartial[split][row] = sum_{cols in split} exp(logit - 2).
// ---------------------------------------------------------------------------
__global__ void __launch_bounds__(256)
k_gemm_lse(const _Float16* __restrict__ reps, float* __restrict__ sPartial) {
    __shared__ _Float16 smem[2][32 * DDIM];   // two 8 KB B-stage buffers

    const int lane = threadIdx.x & 31;
    const int wave = threadIdx.x >> 5;
    const int n    = lane & 15;   // N index within a 16x16 tile
    const int hi   = lane >> 4;   // half-wave select
    const int rowBase  = (blockIdx.x * 8 + wave) * 16;
    const int colStart = blockIdx.y * COLS_PER_SPLIT;

    // Staging: wave w copies B rows {4w .. 4w+3}; each lane moves 16 bytes,
    // so two async b128 issues per wave cover its 1 KB.
    const int stByte0 = (4 * wave + hi) * 256 + n * 16;  // rows 4w, 4w+1
    const int stByte1 = stByte0 + 512;                   // rows 4w+2, 4w+3
    const uint32_t ldsBase = (uint32_t)(uintptr_t)(&smem[0][0]);

    // --- A fragments for this 16-row tile, all K=128 (4 chunks of 32) -----
    // 16-bit A 16x32 layout: lanes 0-15 (hi=0) hold K {0..15},
    // lanes 16-31 (hi=1) hold K {16..31}  (contiguous, matching B)
    h16 a[4];
    const _Float16* aRow = reps + (size_t)(rowBase + n) * DDIM;
    #pragma unroll
    for (int kc = 0; kc < 4; ++kc) {
        union { h16 v; h8 p[2]; } u;
        u.p[0] = *(const h8*)(aRow + kc * 32 + 16 * hi);
        u.p[1] = *(const h8*)(aRow + kc * 32 + 16 * hi + 8);
        a[kc] = u.v;
    }

    float s[8];
    #pragma unroll
    for (int r = 0; r < 8; ++r) s[r] = 0.0f;

    // Prologue: stage columns [colStart, colStart+32) into buffer 0.
    {
        uint32_t d0 = ldsBase + (uint32_t)stByte0;
        uint32_t d1 = ldsBase + (uint32_t)stByte1;
        uint32_t g0 = (uint32_t)(colStart * 256 + stByte0);
        uint32_t g1 = (uint32_t)(colStart * 256 + stByte1);
        asm volatile("global_load_async_to_lds_b128 %0, %1, %2"
                     :: "v"(d0), "v"(g0), "s"(reps) : "memory");
        asm volatile("global_load_async_to_lds_b128 %0, %1, %2"
                     :: "v"(d1), "v"(g1), "s"(reps) : "memory");
    }

    for (int t = 0; t < ITERS_PER_SPLIT; ++t) {
        asm volatile("s_wait_asynccnt 0x0" ::: "memory");
        __syncthreads();
        const int cur = t & 1;

        if (t + 1 < ITERS_PER_SPLIT) {
            const uint32_t bufOff = (uint32_t)((cur ^ 1) * STAGE_BYTES);
            const uint32_t colOff = (uint32_t)((colStart + (t + 1) * 32) * 256);
            uint32_t d0 = ldsBase + bufOff + (uint32_t)stByte0;
            uint32_t d1 = ldsBase + bufOff + (uint32_t)stByte1;
            uint32_t g0 = colOff + (uint32_t)stByte0;
            uint32_t g1 = colOff + (uint32_t)stByte1;
            asm volatile("global_load_async_to_lds_b128 %0, %1, %2"
                         :: "v"(d0), "v"(g0), "s"(reps) : "memory");
            asm volatile("global_load_async_to_lds_b128 %0, %1, %2"
                         :: "v"(d1), "v"(g1), "s"(reps) : "memory");
        }

        // B 32x16 layout: lane holds column N = lane&15; halves K = h + 16*hi
        const _Float16* bLo = &smem[cur][0] + n * DDIM + 16 * hi;        // cols +0..15
        const _Float16* bHi = &smem[cur][0] + (16 + n) * DDIM + 16 * hi; // cols +16..31

        // Four independent accumulator chains (depth 2) across 8 WMMAs.
        f8 acc00 = {}, acc01 = {}, acc10 = {}, acc11 = {};
        acc00 = __builtin_amdgcn_wmma_f32_16x16x32_f16(false, a[0], false,
                    *(const h16*)(bLo + 0 * 32), (short)0, acc00, false, false);
        acc10 = __builtin_amdgcn_wmma_f32_16x16x32_f16(false, a[0], false,
                    *(const h16*)(bHi + 0 * 32), (short)0, acc10, false, false);
        acc01 = __builtin_amdgcn_wmma_f32_16x16x32_f16(false, a[1], false,
                    *(const h16*)(bLo + 1 * 32), (short)0, acc01, false, false);
        acc11 = __builtin_amdgcn_wmma_f32_16x16x32_f16(false, a[1], false,
                    *(const h16*)(bHi + 1 * 32), (short)0, acc11, false, false);
        acc00 = __builtin_amdgcn_wmma_f32_16x16x32_f16(false, a[2], false,
                    *(const h16*)(bLo + 2 * 32), (short)0, acc00, false, false);
        acc10 = __builtin_amdgcn_wmma_f32_16x16x32_f16(false, a[2], false,
                    *(const h16*)(bHi + 2 * 32), (short)0, acc10, false, false);
        acc01 = __builtin_amdgcn_wmma_f32_16x16x32_f16(false, a[3], false,
                    *(const h16*)(bLo + 3 * 32), (short)0, acc01, false, false);
        acc11 = __builtin_amdgcn_wmma_f32_16x16x32_f16(false, a[3], false,
                    *(const h16*)(bHi + 3 * 32), (short)0, acc11, false, false);

        // exp(2a-2) = exp2(a*EXP2C - EXP2C): single pk_fma + v_exp_f32 each.
        #pragma unroll
        for (int r = 0; r < 8; ++r) {
            s[r] += __builtin_amdgcn_exp2f((acc00[r] + acc01[r]) * EXP2C - EXP2C);
            s[r] += __builtin_amdgcn_exp2f((acc10[r] + acc11[r]) * EXP2C - EXP2C);
        }
    }

    // Reduce each row's partial across the 16 lanes of its half-group.
    #pragma unroll
    for (int r = 0; r < 8; ++r) {
        float t = s[r];
        t += __shfl_xor(t, 1);
        t += __shfl_xor(t, 2);
        t += __shfl_xor(t, 4);
        t += __shfl_xor(t, 8);
        if (n == 0)
            sPartial[(size_t)blockIdx.y * TWO_N + rowBase + r + 8 * hi] = t;
    }
}

// ---------------------------------------------------------------------------
// Kernel 3: per-row term = lse - pos_logit.
// One wave per row: exact f32 dots (positive partner AND self) from the f16
// rows; combine the CSPLIT partials, subtract the self term that kernel 2
// included, -> lse = 2 + log(sum - exp(2*selfdot - 2)).
// ---------------------------------------------------------------------------
__global__ void __launch_bounds__(256)
k_rowterm(const _Float16* __restrict__ reps, const float* __restrict__ sPartial,
          float* __restrict__ rowterm) {
    const int lane = threadIdx.x & 31;
    const int wave = threadIdx.x >> 5;
    const int g = blockIdx.x * (blockDim.x >> 5) + wave;
    if (g >= TWO_N) return;
    const int p = (g + NPAIR) & (TWO_N - 1);

    h4 av = ((const h4*)(reps + (size_t)g * DDIM))[lane];
    h4 bv = ((const h4*)(reps + (size_t)p * DDIM))[lane];
    float dotp = 0.0f, dots = 0.0f;
    #pragma unroll
    for (int j = 0; j < 4; ++j) {
        const float x = (float)av[j];
        dotp += x * (float)bv[j];
        dots += x * x;
    }
    dotp += __shfl_xor(dotp, 1);
    dotp += __shfl_xor(dotp, 2);
    dotp += __shfl_xor(dotp, 4);
    dotp += __shfl_xor(dotp, 8);
    dotp += __shfl_xor(dotp, 16);
    dots += __shfl_xor(dots, 1);
    dots += __shfl_xor(dots, 2);
    dots += __shfl_xor(dots, 4);
    dots += __shfl_xor(dots, 8);
    dots += __shfl_xor(dots, 16);

    if (lane == 0) {
        float sum = 0.0f;
        #pragma unroll
        for (int c = 0; c < CSPLIT; ++c) sum += sPartial[(size_t)c * TWO_N + g];
        // remove the self (diagonal) term that kernel 2 included
        sum -= __builtin_amdgcn_exp2f(dots * EXP2C - EXP2C);
        // lse = 2 + log(sum);  pos_logit = 2*dotp
        rowterm[g] = 2.0f + logf(sum) - 2.0f * dotp;
    }
}

// ---------------------------------------------------------------------------
// Kernel 4: mean over the 8192 row terms -> scalar loss.
// ---------------------------------------------------------------------------
__global__ void __launch_bounds__(256)
k_reduce(const float* __restrict__ rowterm, float* __restrict__ out) {
    __shared__ float sm[256];
    float acc = 0.0f;
    for (int i = threadIdx.x; i < TWO_N; i += 256) acc += rowterm[i];
    sm[threadIdx.x] = acc;
    __syncthreads();
    for (int off = 128; off > 0; off >>= 1) {
        if ((int)threadIdx.x < off) sm[threadIdx.x] += sm[threadIdx.x + off];
        __syncthreads();
    }
    if (threadIdx.x == 0) out[0] = sm[0] / (float)TWO_N;
}

// ---------------------------------------------------------------------------
extern "C" void kernel_launch(void* const* d_in, const int* in_sizes, int n_in,
                              void* d_out, int out_size, void* d_ws, size_t ws_size,
                              hipStream_t stream) {
    const float* emb_i = (const float*)d_in[0];
    const float* emb_j = (const float*)d_in[1];
    float* out = (float*)d_out;

    // Workspace layout
    _Float16* reps = (_Float16*)d_ws;                        // 2 MB  (8192x128 f16)
    float* sPartial = (float*)((char*)d_ws +
                               (size_t)TWO_N * DDIM * sizeof(_Float16)); // 256 KB
    float* rowterm = sPartial + (size_t)CSPLIT * TWO_N;      // 32 KB

    k_normalize<<<TWO_N / 8, 256, 0, stream>>>(emb_i, emb_j, reps);
    k_gemm_lse<<<dim3(TWO_N / 128, CSPLIT), 256, 0, stream>>>(reps, sPartial);
    k_rowterm<<<TWO_N / 8, 256, 0, stream>>>(reps, sPartial, rowterm);
    k_reduce<<<1, 256, 0, stream>>>(rowterm, out);
}
